// LSTMModel_8598524526709
// MI455X (gfx1250) — matrix operand order repaired and checked by reading, not compiled, not run on hardware
//
#include <hip/hip_runtime.h>

// LSTM dims (compile-time constants matching the reference)
namespace {
constexpr int Tn  = 512;  // seq len
constexpr int Dn  = 128;  // input dim
constexpr int Hn  = 64;   // hidden dim
constexpr int Gn  = 256;  // 4*H (gates)
constexpr int Bn  = 512;  // batch
constexpr int BCn = 32;   // batch rows per workgroup (2 M-tiles)
}

typedef __bf16  bf16_t;
typedef bf16_t  v16bf __attribute__((ext_vector_type(16)));
typedef float   v8f   __attribute__((ext_vector_type(8)));
typedef unsigned int u32x4 __attribute__((ext_vector_type(4)));
typedef float   f32x4 __attribute__((ext_vector_type(4)));

union FragU { u32x4 q[2]; v16bf v; };
union Pack8 { bf16_t h[8]; u32x4 q; };

__device__ __forceinline__ float sigm(float x) { return 1.0f / (1.0f + __expf(-x)); }
__device__ __forceinline__ float tanha(float x) { return 2.0f / (1.0f + __expf(-2.0f * x)) - 1.0f; }

// Two contiguous 16B LDS chunks -> one bf16 WMMA fragment.
__device__ __forceinline__ v16bf ldfrag(const bf16_t* p0, const bf16_t* p1) {
  FragU f;
  f.q[0] = *(const u32x4*)p0;
  f.q[1] = *(const u32x4*)p1;
  return f.v;
}

// Recurrent GEMM fragment pass: acc[q] += hA(16x64 bf16 in LDS) * W^T tiles.
__device__ __forceinline__ void gemm_h(const bf16_t* sH, const bf16_t* sW,
                                       v8f* acc, int arow, int hi, int ncb) {
#pragma unroll
  for (int kt = 0; kt < 2; ++kt) {
    v16bf a = ldfrag(&sH[arow * Hn + kt * 32 + hi * 8],
                     &sH[arow * Hn + kt * 32 + 16 + hi * 8]);
#pragma unroll
    for (int qq = 0; qq < 4; ++qq) {
      const int nc = ncb + qq * 16;  // gate column (row of W)
      v16bf b = ldfrag(&sW[nc * Hn + kt * 32 + hi * 8],
                       &sW[nc * Hn + kt * 32 + 16 + hi * 8]);
      acc[qq] = __builtin_amdgcn_wmma_f32_16x16x32_bf16(false, a, false, b,
                                                        (short)0, acc[qq], false, false);
    }
  }
}

// Input projection for one timestep: xa[q] = bias + x_t @ Wih0^T tiles.
// Depends only on global x and read-only LDS weights -> software-pipelinable.
__device__ __forceinline__ void xproj(const float* xp, const bf16_t* sWih0,
                                      const float* b0v, v8f* xa, int hi, int ncb) {
#pragma unroll
  for (int q = 0; q < 4; ++q)
#pragma unroll
    for (int j = 0; j < 8; ++j) xa[q][j] = b0v[q];
  __builtin_prefetch(xp + Dn, 0, 0);  // global_prefetch the following timestep row
#pragma unroll
  for (int kt = 0; kt < 4; ++kt) {
    const f32x4* p0 = (const f32x4*)(xp + kt * 32 + hi * 8);
    const f32x4* p1 = (const f32x4*)(xp + kt * 32 + 16 + hi * 8);
    f32x4 f0 = p0[0], f1 = p0[1], f2 = p1[0], f3 = p1[1];
    v16bf a;
    a[0]  = (bf16_t)f0.x; a[1]  = (bf16_t)f0.y; a[2]  = (bf16_t)f0.z; a[3]  = (bf16_t)f0.w;
    a[4]  = (bf16_t)f1.x; a[5]  = (bf16_t)f1.y; a[6]  = (bf16_t)f1.z; a[7]  = (bf16_t)f1.w;
    a[8]  = (bf16_t)f2.x; a[9]  = (bf16_t)f2.y; a[10] = (bf16_t)f2.z; a[11] = (bf16_t)f2.w;
    a[12] = (bf16_t)f3.x; a[13] = (bf16_t)f3.y; a[14] = (bf16_t)f3.z; a[15] = (bf16_t)f3.w;
#pragma unroll
    for (int qq = 0; qq < 4; ++qq) {
      const int nc = ncb + qq * 16;
      v16bf b = ldfrag(&sWih0[nc * Dn + kt * 32 + hi * 8],
                       &sWih0[nc * Dn + kt * 32 + 16 + hi * 8]);
      xa[qq] = __builtin_amdgcn_wmma_f32_16x16x32_bf16(false, a, false, b,
                                                       (short)0, xa[qq], false, false);
    }
  }
}

__device__ __forceinline__ void store_z(float* sZ, const v8f* acc, int mrow, int ncb) {
#pragma unroll
  for (int qq = 0; qq < 4; ++qq)
#pragma unroll
    for (int j = 0; j < 8; ++j)
      sZ[(mrow + j) * Gn + ncb + qq * 16] = acc[qq][j];
}

// LSTM cell elementwise: z (i,f,g,o) -> c,h. c stays in this thread's registers.
template <bool KEEP_F32>
__device__ __forceinline__ void lstm_cell(float* sZ, bf16_t* sH,
                                          float* cst, int er, int ec) {
  Pack8 hp;
#pragma unroll
  for (int u = 0; u < 8; ++u) {
    const int col = ec + u;
    float iv = sZ[er * Gn + col];
    float fv = sZ[er * Gn + 64 + col];
    float gv = sZ[er * Gn + 128 + col];
    float ov = sZ[er * Gn + 192 + col];
    float cc = sigm(fv) * cst[u] + sigm(iv) * tanha(gv);
    cst[u] = cc;
    float hh = sigm(ov) * tanha(cc);
    if (KEEP_F32) sZ[er * Gn + col] = hh;  // f32 copy for the final head (last step)
    hp.h[u] = (bf16_t)hh;
  }
  *(u32x4*)&sH[er * Hn + ec] = hp.q;  // 16B aligned bf16 h store
}

__global__ __launch_bounds__(256, 1)
void lstm2_fused(const float* __restrict__ x,
                 const float* __restrict__ Wih0, const float* __restrict__ Whh0,
                 const float* __restrict__ bih0, const float* __restrict__ bhh0,
                 const float* __restrict__ Wih1, const float* __restrict__ Whh1,
                 const float* __restrict__ bih1, const float* __restrict__ bhh1,
                 const float* __restrict__ Wfc,  const float* __restrict__ bfc,
                 float* __restrict__ out) {
  // ~202 KB LDS per workgroup (CDNA5 WGP allows up to 320 KB)
  __shared__ __align__(16) bf16_t sWih0[Gn * Dn];  // 64 KB bf16
  __shared__ __align__(16) bf16_t sWhh0[Gn * Hn];  // 32 KB
  __shared__ __align__(16) bf16_t sWih1[Gn * Hn];  // 32 KB
  __shared__ __align__(16) bf16_t sWhh1[Gn * Hn];  // 32 KB
  __shared__ __align__(16) float  sZ[BCn * Gn];    // 32 KB gate pre-activations
  __shared__ __align__(16) bf16_t sH0[BCn * Hn];   // 4 KB bf16 h (layer 0)
  __shared__ __align__(16) bf16_t sH1[BCn * Hn];   // 4 KB bf16 h (layer 1)
  __shared__ float sB0[Gn];
  __shared__ float sB1[Gn];

  const int tid = (int)threadIdx.x;
  const int b0  = (int)blockIdx.x * BCn;
  const int w    = tid >> 5;
  const int lane = tid & 31;
  const int mw = w >> 2;        // M-tile (0..1)
  const int gw = w & 3;         // N-group (0..3) -> N-tiles {4g..4g+3}
  const int lr = lane & 15;
  const int hi = lane >> 4;
  const int ncb  = gw * 64 + lr;       // base gate column for this lane
  const int arow = mw * 16 + lr;       // A-fragment row in LDS h
  const int mrow = mw * 16 + hi * 8;   // C/D base row

  // ---- one-time: convert weights to bf16 into LDS, combine biases, zero state
  for (int i = tid; i < Gn * Dn; i += 256) sWih0[i] = (bf16_t)Wih0[i];
  for (int i = tid; i < Gn * Hn; i += 256) {
    sWhh0[i] = (bf16_t)Whh0[i];
    sWih1[i] = (bf16_t)Wih1[i];
    sWhh1[i] = (bf16_t)Whh1[i];
  }
  sB0[tid] = bih0[tid] + bhh0[tid];
  sB1[tid] = bih1[tid] + bhh1[tid];
  for (int i = tid; i < BCn * Hn; i += 256) {
    sH0[i] = (bf16_t)0.0f;
    sH1[i] = (bf16_t)0.0f;
  }
  __syncthreads();

  float b0v[4], b1v[4];
#pragma unroll
  for (int q = 0; q < 4; ++q) {
    b0v[q] = sB0[ncb + q * 16];
    b1v[q] = sB1[ncb + q * 16];
  }

  float c0v[8], c1v[8];
#pragma unroll
  for (int u = 0; u < 8; ++u) { c0v[u] = 0.0f; c1v[u] = 0.0f; }

  const int er = tid >> 3;        // elementwise row (0..31)
  const int ec = (tid & 7) * 8;   // elementwise col base
  const float* xrow = x + (size_t)(b0 + mw * 16 + lr) * Tn * Dn;

  // prologue: input projection for t=0
  v8f xacc[4];
  xproj(xrow, sWih0, b0v, xacc, hi, ncb);

  for (int t = 0; t < Tn; ++t) {
    // ===== phase A: z0 = xacc(t) + h0 @ Whh0^T  (only 8 h-dependent WMMAs) =====
    gemm_h(sH0, sWhh0, xacc, arow, hi, ncb);
    store_z(sZ, xacc, mrow, ncb);
    __syncthreads();

    // ===== phase B: layer-0 cell =====
    lstm_cell<false>(sZ, sH0, c0v, er, ec);
    __syncthreads();

    // ===== phase C: z1 = b1 + h0 @ Wih1^T + h1 @ Whh1^T =====
    v8f acc[4];
#pragma unroll
    for (int q = 0; q < 4; ++q)
#pragma unroll
      for (int j = 0; j < 8; ++j) acc[q][j] = b1v[q];
    gemm_h(sH0, sWih1, acc, arow, hi, ncb);
    gemm_h(sH1, sWhh1, acc, arow, hi, ncb);
    store_z(sZ, acc, mrow, ncb);

    // software pipeline: next step's input projection (h-independent; only
    // reads global x and read-only LDS weights, so it is hazard-free here and
    // its global-load latency overlaps the layer-1 cell phase + barriers).
    const int tt = (t + 1) & (Tn - 1);
    xproj(xrow + (size_t)tt * Dn, sWih0, b0v, xacc, hi, ncb);
    __syncthreads();

    // ===== phase D: layer-1 cell (keep f32 h for the head) =====
    lstm_cell<true>(sZ, sH1, c1v, er, ec);
    __syncthreads();
  }

  // ===== final linear head: out[b] = h1_last[b,:] . Wfc + bfc =====
  if (tid < BCn) {
    float s = bfc[0];
#pragma unroll 8
    for (int c = 0; c < Hn; ++c) s += sZ[tid * Gn + c] * Wfc[c];
    out[b0 + tid] = s;
  }
}

extern "C" void kernel_launch(void* const* d_in, const int* in_sizes, int n_in,
                              void* d_out, int out_size, void* d_ws, size_t ws_size,
                              hipStream_t stream) {
  (void)in_sizes; (void)n_in; (void)d_ws; (void)ws_size; (void)out_size;
  const float* x    = (const float*)d_in[0];
  const float* Wih0 = (const float*)d_in[1];
  const float* Whh0 = (const float*)d_in[2];
  const float* bih0 = (const float*)d_in[3];
  const float* bhh0 = (const float*)d_in[4];
  const float* Wih1 = (const float*)d_in[5];
  const float* Whh1 = (const float*)d_in[6];
  const float* bih1 = (const float*)d_in[7];
  const float* bhh1 = (const float*)d_in[8];
  const float* Wfc  = (const float*)d_in[9];
  const float* bfc  = (const float*)d_in[10];
  float* out = (float*)d_out;

  lstm2_fused<<<dim3(Bn / BCn), dim3(256), 0, stream>>>(
      x, Wih0, Whh0, bih0, bhh0, Wih1, Whh1, bih1, bhh1, Wfc, bfc, out);
}
